// BDH_31233002176612
// MI455X (gfx1250) — matrix operand order repaired and checked by reading, not compiled, not run on hardware
//
#include <hip/hip_runtime.h>
#include <stdint.h>
#include <math.h>

// ---- model constants ----
#define Db     768
#define NHb    12
#define HDb    512
#define NHN    6144     // NH * HD
#define Tb     1024
#define Bb     2
#define ROWS   2048     // B * T
#define VOCABb 32000
#define Lb     4
#define KEEPb  921      // int(6144 * 0.15)
#define TWO_PI 6.283185307179586f

typedef __attribute__((ext_vector_type(16))) __bf16 v16bf;
typedef __attribute__((ext_vector_type(8)))  float  v8f;
typedef __attribute__((ext_vector_type(4)))  int    v4i;

__device__ __forceinline__ uint16_t f2bf(float f) {
  union { float f; uint32_t u; } x; x.f = f;
  uint32_t r = (x.u + 0x7FFFu + ((x.u >> 16) & 1u)) >> 16;
  return (uint16_t)r;
}
__device__ __forceinline__ float bf2f(uint16_t b) {
  union { uint32_t u; float f; } x; x.u = ((uint32_t)b) << 16;
  return x.f;
}

// CDNA5 async global->LDS copy path (ASYNCcnt-tracked, no VGPR round trip).
#if __has_builtin(__builtin_amdgcn_global_load_async_to_lds_b128) && \
    __has_builtin(__builtin_amdgcn_s_wait_asynccnt)
#define USE_ASYNC_COPY 1
#else
#define USE_ASYNC_COPY 0
#endif

#if USE_ASYNC_COPY
typedef __attribute__((address_space(1))) v4i* glb_v4i_p;
typedef __attribute__((address_space(3))) v4i* lds_v4i_p;
__device__ __forceinline__ void async_cp_b128(const uint16_t* g, uint16_t* l) {
  __builtin_amdgcn_global_load_async_to_lds_b128(
      (glb_v4i_p)(g), (lds_v4i_p)(l), 0, 0);
}
#endif

// =====================================================================
// Batched WMMA GEMM:  C[M,N] (f32) = A[M,K] (bf16, row-major) * BT[N,K]^T
// Block tile 128x128, K-step 64, 8 waves (2M x 4N), wave tile 64x32.
// Double-buffered LDS filled with GLOBAL_LOAD_ASYNC_TO_LDS_B128.
// =====================================================================
#define BM 128
#define BN 128
#define BK 64
#define LDP 72   // padded LDS row stride (elements): 144B spreads banks

__global__ __launch_bounds__(256) void k_gemm_bf16(
    const uint16_t* __restrict__ A, const uint16_t* __restrict__ BT,
    float* __restrict__ C, int M, int Nn, int K,
    long long strideA, long long strideB, long long strideC)
{
  __shared__ __align__(16) uint16_t As[2][BM * LDP];
  __shared__ __align__(16) uint16_t Bs[2][BN * LDP];

  const int batch = blockIdx.z;
  A  += (long long)batch * strideA;
  BT += (long long)batch * strideB;
  C  += (long long)batch * strideC;

  const int bm  = blockIdx.y * BM;
  const int bn  = blockIdx.x * BN;
  const int tid = threadIdx.x;
  const int lane = tid & 31;
  const int wid  = tid >> 5;
  const int wm = (wid & 1) * 64;   // wave M offset in block tile
  const int wn = (wid >> 1) * 32;  // wave N offset in block tile
  const int m  = lane & 15;        // fragment row/col index
  const int h  = lane >> 4;        // lane half

  // tile-copy mapping: 2 threads per row, 32 cols (64B = 4x b128) each
  const int crow = tid >> 1;        // 0..127
  const int ccol = (tid & 1) * 32;  // 0 or 32

  v8f acc[4][2] = {};

  const int ktiles = K / BK;

  auto issue_async = [&](int it, int buf) {
#if USE_ASYNC_COPY
    const uint16_t* ga = &A [(long long)(bm + crow) * K + it * BK + ccol];
    const uint16_t* gb = &BT[(long long)(bn + crow) * K + it * BK + ccol];
    uint16_t* la = &As[buf][crow * LDP + ccol];
    uint16_t* lb = &Bs[buf][crow * LDP + ccol];
#pragma unroll
    for (int q = 0; q < 4; ++q) async_cp_b128(ga + q * 8, la + q * 8);
#pragma unroll
    for (int q = 0; q < 4; ++q) async_cp_b128(gb + q * 8, lb + q * 8);
#else
    (void)it; (void)buf;
#endif
  };

  auto compute = [&](int buf) {
#pragma unroll
    for (int kk = 0; kk < BK; kk += 32) {
      union frag { v16bf v; uint4 q[2]; };
      frag af[4], bfr[2];
#pragma unroll
      for (int mi = 0; mi < 4; ++mi) {
        const uint16_t* base = &As[buf][(wm + mi * 16 + m) * LDP + kk];
        af[mi].q[0] = *(const uint4*)(base + h * 8);
        af[mi].q[1] = *(const uint4*)(base + 16 + h * 8);
      }
#pragma unroll
      for (int ni = 0; ni < 2; ++ni) {
        const uint16_t* base = &Bs[buf][(wn + ni * 16 + m) * LDP + kk];
        bfr[ni].q[0] = *(const uint4*)(base + h * 8);
        bfr[ni].q[1] = *(const uint4*)(base + 16 + h * 8);
      }
#pragma unroll
      for (int mi = 0; mi < 4; ++mi)
#pragma unroll
        for (int ni = 0; ni < 2; ++ni)
          acc[mi][ni] = __builtin_amdgcn_wmma_f32_16x16x32_bf16(
              false, af[mi].v, false, bfr[ni].v, (short)0, acc[mi][ni],
              false, false);
    }
  };

#if USE_ASYNC_COPY
  issue_async(0, 0);
  for (int it = 0; it < ktiles; ++it) {
    const int cur = it & 1;
    if (it + 1 < ktiles) {
      issue_async(it + 1, cur ^ 1);           // prefetch next tile
      __builtin_amdgcn_s_wait_asynccnt(8);    // current tile's 8 ops done
    } else {
      __builtin_amdgcn_s_wait_asynccnt(0);
    }
    __syncthreads();
    compute(cur);
    __syncthreads();   // all waves done reading before buffer reuse
  }
#else
  for (int it = 0; it < ktiles; ++it) {
    uint4 ra[4], rb[4];
    const uint16_t* ga = &A [(long long)(bm + crow) * K + it * BK + ccol];
    const uint16_t* gb = &BT[(long long)(bn + crow) * K + it * BK + ccol];
#pragma unroll
    for (int q = 0; q < 4; ++q) ra[q] = *(const uint4*)(ga + q * 8);
#pragma unroll
    for (int q = 0; q < 4; ++q) rb[q] = *(const uint4*)(gb + q * 8);
    __syncthreads();
#pragma unroll
    for (int q = 0; q < 4; ++q) {
      *(uint4*)&As[0][crow * LDP + ccol + q * 8] = ra[q];
      *(uint4*)&Bs[0][crow * LDP + ccol + q * 8] = rb[q];
    }
    __syncthreads();
    compute(0);
  }
#endif

  // C/D layout: lane half h holds rows h*8..h*8+7, col = lane&15
  const int row0 = bm + wm + h * 8;
  const int col0 = bn + wn + m;
#pragma unroll
  for (int mi = 0; mi < 4; ++mi)
#pragma unroll
    for (int ni = 0; ni < 2; ++ni)
#pragma unroll
      for (int r = 0; r < 8; ++r)
        C[(long long)(row0 + mi * 16 + r) * Nn + col0 + ni * 16] =
            acc[mi][ni][r];
}

// =====================================================================
// Weight prep kernels
// =====================================================================
__global__ __launch_bounds__(256) void k_cvt_bf16(
    const float* __restrict__ s, uint16_t* __restrict__ d, long long n) {
  long long i = (long long)blockIdx.x * 256 + threadIdx.x;
  if (i < n) d[i] = f2bf(s[i]);
}

// (R,C) f32 -> (C,R) bf16
__global__ __launch_bounds__(256) void k_transpose_bf16(
    const float* __restrict__ s, uint16_t* __restrict__ d, int R, int C) {
  long long i = (long long)blockIdx.x * 256 + threadIdx.x;
  if (i >= (long long)R * C) return;
  int r = (int)(i / C), c = (int)(i % C);
  d[(long long)c * R + r] = f2bf(s[i]);
}

// encoder (NH, D, HD) f32 -> BT layout (NH*HD, D) bf16
__global__ __launch_bounds__(256) void k_enc_to_bt(
    const float* __restrict__ s, uint16_t* __restrict__ d) {
  long long i = (long long)blockIdx.x * 256 + threadIdx.x;
  if (i >= (long long)NHb * Db * HDb) return;
  int hh = (int)(i / ((long long)Db * HDb));
  long long rem = i % ((long long)Db * HDb);
  int dd = (int)(rem / HDb);
  int nn = (int)(rem % HDb);
  d[((long long)(hh * HDb + nn)) * Db + dd] = f2bf(s[i]);
}

// =====================================================================
// x = layernorm(embed[idx]); also bf16 copy
// =====================================================================
__global__ __launch_bounds__(256) void k_embed_ln(
    const int* __restrict__ idx, const float* __restrict__ emb,
    const float* __restrict__ w, const float* __restrict__ b,
    float* __restrict__ xf, uint16_t* __restrict__ xb) {
  __shared__ float red[256];
  const int row = blockIdx.x, tid = threadIdx.x;
  const float* e = emb + (long long)idx[row] * Db;
  float s = 0.f, s2 = 0.f;
  for (int j = tid; j < Db; j += 256) { float v = e[j]; s += v; s2 += v * v; }
  red[tid] = s; __syncthreads();
  for (int st = 128; st; st >>= 1) { if (tid < st) red[tid] += red[tid + st]; __syncthreads(); }
  float mu = red[0] / Db; __syncthreads();
  red[tid] = s2; __syncthreads();
  for (int st = 128; st; st >>= 1) { if (tid < st) red[tid] += red[tid + st]; __syncthreads(); }
  float var = red[0] / Db - mu * mu;
  float rs = rsqrtf(var + 1e-5f);
  for (int j = tid; j < Db; j += 256) {
    float v = (e[j] - mu) * rs * w[j] + b[j];
    xf[(long long)row * Db + j] = v;
    xb[(long long)row * Db + j] = f2bf(v);
  }
}

// =====================================================================
// layernorm -> relu -> kWTA (binary-search top-k threshold) -> bf16 out
// mode 0: out_lin[row, j]   mode 1: out_vt[(b*NH+h)*HD+n][T] transposed V
// =====================================================================
__global__ __launch_bounds__(256) void k_ln_relu_kwta(
    const float* __restrict__ pre, const float* __restrict__ w,
    const float* __restrict__ b, uint16_t* __restrict__ out_lin,
    uint16_t* __restrict__ out_vt, int mode) {
  __shared__ float rowbuf[NHN];
  __shared__ float red[256];
  __shared__ int cnt;
  const int row = blockIdx.x, tid = threadIdx.x;
  const float* src = pre + (long long)row * NHN;

  float s = 0.f, s2 = 0.f;
  for (int j = tid; j < NHN; j += 256) {
    float v = src[j]; rowbuf[j] = v; s += v; s2 += v * v;
  }
  red[tid] = s; __syncthreads();
  for (int st = 128; st; st >>= 1) { if (tid < st) red[tid] += red[tid + st]; __syncthreads(); }
  float mu = red[0] / NHN; __syncthreads();
  red[tid] = s2; __syncthreads();
  for (int st = 128; st; st >>= 1) { if (tid < st) red[tid] += red[tid + st]; __syncthreads(); }
  float var = red[0] / NHN - mu * mu;
  float rs = rsqrtf(var + 1e-5f);
  __syncthreads();

  float mx = 0.f;
  for (int j = tid; j < NHN; j += 256) {
    float v = (rowbuf[j] - mu) * rs * w[j] + b[j];
    v = v > 0.f ? v : 0.f;
    rowbuf[j] = v;
    mx = fmaxf(mx, v);
  }
  __syncthreads();
  red[tid] = mx; __syncthreads();
  for (int st = 128; st; st >>= 1) { if (tid < st) red[tid] = fmaxf(red[tid], red[tid + st]); __syncthreads(); }
  mx = red[0]; __syncthreads();

  float lo = 0.f, hi = mx;
  for (int it = 0; it < 22; ++it) {
    float mid = 0.5f * (lo + hi);
    if (tid == 0) cnt = 0;
    __syncthreads();
    int c = 0;
    for (int j = tid; j < NHN; j += 256) c += (rowbuf[j] >= mid) ? 1 : 0;
    atomicAdd(&cnt, c);
    __syncthreads();
    int total = cnt;
    __syncthreads();
    if (total >= KEEPb) lo = mid; else hi = mid;
  }
  const float thr = lo;

  if (mode == 0) {
    for (int j = tid; j < NHN; j += 256) {
      float v = rowbuf[j];
      out_lin[(long long)row * NHN + j] = f2bf(v >= thr ? v : 0.f);
    }
  } else {
    const int bq = row / Tb, t = row % Tb;
    for (int j = tid; j < NHN; j += 256) {
      float v = rowbuf[j];
      uint16_t bv = f2bf(v >= thr ? v : 0.f);
      int hh = j >> 9, nn = j & 511;
      out_vt[(((long long)(bq * NHb + hh)) * HDb + nn) * Tb + t] = bv;
    }
  }
}

// =====================================================================
// RoPE: q (b,t,h*HD+n) bf16 -> qr (b,h,t,n) bf16
// =====================================================================
__global__ __launch_bounds__(256) void k_rope(
    const uint16_t* __restrict__ q, uint16_t* __restrict__ qr) {
  long long i = (long long)blockIdx.x * 256 + threadIdx.x;  // pair index
  const long long total = (long long)Bb * NHb * Tb * (HDb / 2);
  if (i >= total) return;
  int np = (int)(i % (HDb / 2)); long long r = i / (HDb / 2);
  int t  = (int)(r % Tb); r /= Tb;
  int hh = (int)(r % NHb);
  int bq = (int)(r / NHb);
  int n  = np * 2;
  long long inoff = ((long long)(bq * Tb + t)) * NHN + hh * HDb + n;
  float x0 = bf2f(q[inoff]);
  float x1 = bf2f(q[inoff + 1]);
  float freq = 1.0f / (powf(10000.0f, (float)n / (float)HDb) * TWO_PI);
  float ph = (float)t * freq;
  float ang = (ph - floorf(ph)) * TWO_PI;
  float c = __cosf(ang), sn = __sinf(ang);
  float o0 = x0 * c - x1 * sn;
  float o1 = x1 * c + x0 * sn;
  long long outoff = (((long long)(bq * NHb + hh)) * Tb + t) * HDb + n;
  qr[outoff]     = f2bf(o0);
  qr[outoff + 1] = f2bf(o1);
}

// =====================================================================
// causal softmax over S rows (length T), writes bf16 att
// =====================================================================
__global__ __launch_bounds__(256) void k_softmax_causal(
    const float* __restrict__ S, uint16_t* __restrict__ att) {
  __shared__ float buf[Tb];
  __shared__ float red[256];
  const int row = blockIdx.x, tid = threadIdx.x;
  const int t = row % Tb;
  const float scale = 0.044194173824159216f;  // 1/sqrt(512)
  const float* src = S + (long long)row * Tb;

  float mx = -3.0e38f;
  for (int j = tid; j < Tb; j += 256) {
    float v = (j <= t) ? src[j] * scale : -3.0e38f;
    buf[j] = v;
    mx = fmaxf(mx, v);
  }
  __syncthreads();
  red[tid] = mx; __syncthreads();
  for (int st = 128; st; st >>= 1) { if (tid < st) red[tid] = fmaxf(red[tid], red[tid + st]); __syncthreads(); }
  mx = red[0]; __syncthreads();

  float s = 0.f;
  for (int j = tid; j < Tb; j += 256) {
    float e = (j <= t) ? __expf(buf[j] - mx) : 0.f;
    buf[j] = e;
    s += e;
  }
  __syncthreads();
  red[tid] = s; __syncthreads();
  for (int st = 128; st; st >>= 1) { if (tid < st) red[tid] += red[tid + st]; __syncthreads(); }
  s = red[0];
  const float inv = 1.0f / s;
  for (int j = tid; j < Tb; j += 256)
    att[(long long)row * Tb + j] = f2bf(buf[j] * inv);
}

// =====================================================================
// y (b,h,t,n) f32 -> ycat (b,t, h*HD+n) bf16
// =====================================================================
__global__ __launch_bounds__(256) void k_ycat(
    const float* __restrict__ y, uint16_t* __restrict__ yc) {
  long long i = (long long)blockIdx.x * 256 + threadIdx.x;
  if (i >= (long long)Bb * Tb * NHN) return;
  int j = (int)(i % NHN); long long r = i / NHN;
  int t  = (int)(r % Tb);
  int bq = (int)(r / Tb);
  int hh = j >> 9, nn = j & 511;
  yc[i] = f2bf(y[(((long long)(bq * NHb + hh)) * Tb + t) * HDb + nn]);
}

// =====================================================================
// x = residual + layernorm(ydec + bias); also bf16 copy (in-place on x)
// =====================================================================
__global__ __launch_bounds__(256) void k_ln_residual(
    const float* __restrict__ ydec, const float* __restrict__ dbias,
    const float* __restrict__ w, const float* __restrict__ b,
    float* __restrict__ x, uint16_t* __restrict__ xb) {
  __shared__ float red[256];
  __shared__ float rbuf[Db];
  const int row = blockIdx.x, tid = threadIdx.x;
  float s = 0.f, s2 = 0.f;
  for (int j = tid; j < Db; j += 256) {
    float v = ydec[(long long)row * Db + j] + dbias[j];
    rbuf[j] = v; s += v; s2 += v * v;
  }
  red[tid] = s; __syncthreads();
  for (int st = 128; st; st >>= 1) { if (tid < st) red[tid] += red[tid + st]; __syncthreads(); }
  float mu = red[0] / Db; __syncthreads();
  red[tid] = s2; __syncthreads();
  for (int st = 128; st; st >>= 1) { if (tid < st) red[tid] += red[tid + st]; __syncthreads(); }
  float var = red[0] / Db - mu * mu;
  float rs = rsqrtf(var + 1e-5f);
  for (int j = tid; j < Db; j += 256) {
    float v = (rbuf[j] - mu) * rs * w[j] + b[j];
    float o = x[(long long)row * Db + j] + v;
    x[(long long)row * Db + j] = o;
    xb[(long long)row * Db + j] = f2bf(o);
  }
}

// =====================================================================
// host launch
// =====================================================================
static inline int nblk(long long n) { return (int)((n + 255) / 256); }

extern "C" void kernel_launch(void* const* d_in, const int* in_sizes, int n_in,
                              void* d_out, int out_size, void* d_ws, size_t ws_size,
                              hipStream_t stream) {
  (void)in_sizes; (void)n_in; (void)out_size; (void)ws_size;

  const int*   idx     = (const int*)d_in[0];
  const float* emb     = (const float*)d_in[1];
  const float* ln_in_w = (const float*)d_in[2];
  const float* ln_in_b = (const float*)d_in[3];
  const float* enc     = (const float*)d_in[4];
  const float* encv    = (const float*)d_in[5];
  const float* lnq_w   = (const float*)d_in[6];
  const float* lnq_b   = (const float*)d_in[7];
  const float* lnv_w   = (const float*)d_in[8];
  const float* lnv_b   = (const float*)d_in[9];
  const float* dec_w   = (const float*)d_in[10];
  const float* dec_b   = (const float*)d_in[11];
  const float* lno_w   = (const float*)d_in[12];
  const float* lno_b   = (const float*)d_in[13];
  const float* lmh     = (const float*)d_in[14];
  float* logits        = (float*)d_out;

  char* ws = (char*)d_ws;
  size_t off = 0;
  auto alloc = [&](size_t bytes) -> void* {
    off = (off + 255) & ~(size_t)255;
    void* p = ws + off;
    off += bytes;
    return p;
  };

  float*    xf    = (float*)   alloc((size_t)ROWS * Db * 4);
  uint16_t* xb    = (uint16_t*)alloc((size_t)ROWS * Db * 2);
  uint16_t* wencT = (uint16_t*)alloc((size_t)NHN * Db * 2);
  uint16_t* wencvT= (uint16_t*)alloc((size_t)NHN * Db * 2);
  uint16_t* wdecT = (uint16_t*)alloc((size_t)Db * NHN * 2);
  uint16_t* lmhb  = (uint16_t*)alloc((size_t)VOCABb * Db * 2);
  float*    pre   = (float*)   alloc((size_t)ROWS * NHN * 4);
  uint16_t* qb    = (uint16_t*)alloc((size_t)ROWS * NHN * 2);
  uint16_t* vt    = (uint16_t*)alloc((size_t)ROWS * NHN * 2);
  uint16_t* qr    = (uint16_t*)alloc((size_t)ROWS * NHN * 2);
  float*    Sbuf  = (float*)   alloc((size_t)Bb * NHb * Tb * Tb * 4);
  uint16_t* attb  = (uint16_t*)alloc((size_t)Bb * NHb * Tb * Tb * 2);
  float*    ybuf  = (float*)   alloc((size_t)Bb * NHb * Tb * HDb * 4);
  uint16_t* ycb   = (uint16_t*)alloc((size_t)ROWS * NHN * 2);
  float*    ydec  = (float*)   alloc((size_t)ROWS * Db * 4);

  // ---- weight prep ----
  k_enc_to_bt<<<nblk((long long)NHb * Db * HDb), 256, 0, stream>>>(enc, wencT);
  k_enc_to_bt<<<nblk((long long)NHb * Db * HDb), 256, 0, stream>>>(encv, wencvT);
  k_transpose_bf16<<<nblk((long long)NHN * Db), 256, 0, stream>>>(dec_w, wdecT, NHN, Db);
  k_cvt_bf16<<<nblk((long long)VOCABb * Db), 256, 0, stream>>>(lmh, lmhb, (long long)VOCABb * Db);

  // ---- embedding + input layernorm ----
  k_embed_ln<<<ROWS, 256, 0, stream>>>(idx, emb, ln_in_w, ln_in_b, xf, xb);

  const long long sQK = (long long)Tb * HDb;   // per-(b,h) q/k/v stride
  const long long sSS = (long long)Tb * Tb;    // per-(b,h) score stride

  for (int i = 0; i < Lb; ++i) {
    // q_pre = x @ W_enc
    k_gemm_bf16<<<dim3(NHN / BN, ROWS / BM, 1), 256, 0, stream>>>(
        xb, wencT, pre, ROWS, NHN, Db, 0, 0, 0);
    k_ln_relu_kwta<<<ROWS, 256, 0, stream>>>(
        pre, lnq_w + (long long)i * NHN, lnq_b + (long long)i * NHN,
        qb, (uint16_t*)nullptr, 0);

    // v_pre = x @ W_enc_v  (reuses pre)
    k_gemm_bf16<<<dim3(NHN / BN, ROWS / BM, 1), 256, 0, stream>>>(
        xb, wencvT, pre, ROWS, NHN, Db, 0, 0, 0);
    k_ln_relu_kwta<<<ROWS, 256, 0, stream>>>(
        pre, lnv_w + (long long)i * NHN, lnv_b + (long long)i * NHN,
        (uint16_t*)nullptr, vt, 1);

    // rope(q) -> qr (b,h,t,n)
    k_rope<<<nblk((long long)Bb * NHb * Tb * (HDb / 2)), 256, 0, stream>>>(qb, qr);

    // S = qr @ qr^T  per (b,h)
    k_gemm_bf16<<<dim3(Tb / BN, Tb / BM, Bb * NHb), 256, 0, stream>>>(
        qr, qr, Sbuf, Tb, Tb, HDb, sQK, sQK, sSS);

    // causal softmax
    k_softmax_causal<<<Bb * NHb * Tb, 256, 0, stream>>>(Sbuf, attb);

    // y = att @ v  per (b,h)   (BT = vt, which is v transposed per head)
    k_gemm_bf16<<<dim3(HDb / BN, Tb / BM, Bb * NHb), 256, 0, stream>>>(
        attb, vt, ybuf, Tb, HDb, Tb, sSS, sQK, sQK);

    // concat heads -> (b,t, NH*HD) bf16
    k_ycat<<<nblk((long long)ROWS * NHN), 256, 0, stream>>>(ybuf, ycb);

    // ydec = ycat @ W_dec
    k_gemm_bf16<<<dim3(Db / BN, ROWS / BM, 1), 256, 0, stream>>>(
        ycb, wdecT, ydec, ROWS, Db, NHN, 0, 0, 0);

    // x = residual + layernorm(ydec + b)
    k_ln_residual<<<ROWS, 256, 0, stream>>>(ydec, dec_b, lno_w, lno_b, xf, xb);
  }

  // logits = x @ lm_head^T
  k_gemm_bf16<<<dim3(VOCABb / BN, ROWS / BM, 1), 256, 0, stream>>>(
      xb, lmhb, logits, ROWS, VOCABb, Db, 0, 0, 0);
}
